// AdvancedNER_14809047236796
// MI455X (gfx1250) — matrix operand order repaired and checked by reading, not compiled, not run on hardware
//
#include <hip/hip_runtime.h>
#include <math.h>

typedef __attribute__((ext_vector_type(16))) _Float16 v16h;
typedef __attribute__((ext_vector_type(8)))  _Float16 v8h;
typedef __attribute__((ext_vector_type(8)))  float    v8f;
typedef __attribute__((ext_vector_type(4)))  float    f4;

#define BB 64
#define SS 512
#define EE 768
#define TT 16
#define ROWS (BB*SS)   // 32768

__device__ __forceinline__ v8f wmma16(v16h a, v16h b, v8f c) {
    // D(16x16,f32) = A(16x32,f16) x B(32x16,f16) + C
    return __builtin_amdgcn_wmma_f32_16x16x32_f16(false, a, false, b, (short)0, c, false, false);
}

// A-fragment (16x32 f16): half j of lane (g = lane>>4):
//   j=0..15 -> K = kbase + g*16 + j
__device__ __forceinline__ v16h pack_a_from_f32(const float* row, int kbase, int g) {
    const f4* p0 = (const f4*)(row + kbase + g*16);
    const f4* p1 = (const f4*)(row + kbase + g*16 + 8);
    f4 a0 = p0[0], a1 = p0[1], b0 = p1[0], b1 = p1[1];
    v16h a;
#pragma unroll
    for (int t = 0; t < 4; t++) {
        a[t]    = (_Float16)a0[t];
        a[4+t]  = (_Float16)a1[t];
        a[8+t]  = (_Float16)b0[t];
        a[12+t] = (_Float16)b1[t];
    }
    return a;
}

// B-fragment (32x16 f16): lane holds column n, halves j -> K = g*16 + j (16 consecutive)
__device__ __forceinline__ v16h pack_b16(const float* p) {
    const f4* q = (const f4*)p;
    f4 c0 = q[0], c1 = q[1], c2 = q[2], c3 = q[3];
    v16h b;
#pragma unroll
    for (int t = 0; t < 4; t++) {
        b[t]    = (_Float16)c0[t];
        b[4+t]  = (_Float16)c1[t];
        b[8+t]  = (_Float16)c2[t];
        b[12+t] = (_Float16)c3[t];
    }
    return b;
}

// ---------------------------------------------------------------------------
// Kernel 1: xp = bert_embed @ Wih^T + bih for both directions (N = 384 combined)
// M=32768, N=384, K=768.  Memory-bound: ~100MB read -> ~5us at 23.3 TB/s.
// ---------------------------------------------------------------------------
__global__ void k_gates(const float* __restrict__ x, const float* __restrict__ Wf,
                        const float* __restrict__ Wb, const float* __restrict__ bf,
                        const float* __restrict__ bb, float* __restrict__ xp) {
    int wave = blockIdx.x * 4 + (threadIdx.x >> 5);
    int lane = threadIdx.x & 31, hl = lane & 15, g = lane >> 4;
    int mt = wave / 24, nt = wave % 24;
    const float* W    = (nt < 12) ? Wf : Wb;
    const float* bias = (nt < 12) ? bf : bb;
    int nloc = (nt % 12) * 16 + hl;                  // column within 0..191
    const float* arow = x + (long)(mt*16 + hl) * EE;
    const float* brow = W + (long)nloc * EE;
    v8f c = {};
    for (int kb = 0; kb < EE; kb += 32) {
        v16h a = pack_a_from_f32(arow, kb, g);
        v16h b = pack_b16(brow + kb + g*16);
        c = wmma16(a, b, c);
    }
    float bv = bias[nloc];
    int ncol = nt*16 + hl;
#pragma unroll
    for (int r = 0; r < 8; r++) {
        int m = mt*16 + g*8 + r;
        xp[(long)m*384 + ncol] = c[r] + bv;
    }
}

// ---------------------------------------------------------------------------
// Kernel 2: GRU scan. 8 workgroups = (4 batch-groups x 2 directions),
// 128 threads. Per step: gh = h(16x64) @ Whh^T(64x192) via 12 WMMA tiles
// (Whh fragments resident in VGPRs), then elementwise gates. h staged in LDS
// already in WMMA A-fragment order.
// ---------------------------------------------------------------------------
__global__ void __launch_bounds__(128) k_gru(
        const float* __restrict__ xp,
        const float* __restrict__ Whh_f, const float* __restrict__ Whh_b,
        const float* __restrict__ bhh_f, const float* __restrict__ bhh_b,
        float* __restrict__ lstm) {
    __shared__ v16h  hA[2][32];        // h as A-fragments: [kfrag][lane] (2 KB)
    __shared__ float gh[16][192];      // gate pre-activations (12 KB)

    int dir = blockIdx.x & 1, bg = blockIdx.x >> 1;
    const float* Whh = dir ? Whh_b : Whh_f;
    const float* bhh = dir ? bhh_b : bhh_f;
    int tid = threadIdx.x, wave = tid >> 5, lane = tid & 31, hl = lane & 15, g = lane >> 4;

    // Preload Whh as B-fragments: gh[b][n] = sum_k h[b][k]*Whh[n][k] -> B[k][n]=Whh[n*64+k]
    v16h bw[3][2];
#pragma unroll
    for (int tt = 0; tt < 3; tt++) {
        int n = (wave*3 + tt)*16 + hl;
#pragma unroll
        for (int kf = 0; kf < 2; kf++)
            bw[tt][kf] = pack_b16(Whh + n*64 + kf*32 + g*16);
    }

    int b_loc = tid & 15, k0 = (tid >> 4) * 8;
    float h[8], br[8], bz[8], bn[8];
#pragma unroll
    for (int i = 0; i < 8; i++) {
        h[i] = 0.f;
        br[i] = bhh[k0+i]; bz[i] = bhh[64+k0+i]; bn[i] = bhh[128+k0+i];
    }
    {   // zero h fragments
        _Float16* ha = (_Float16*)hA;
        for (int i = tid; i < 1024; i += 128) ha[i] = (_Float16)0.f;
    }
    __syncthreads();

    int Bidx = bg*16 + b_loc;
    for (int step = 0; step < SS; step++) {
        int s = dir ? (SS-1-step) : step;
        // phase 1: matrix part (all 4 waves, 3 tiles each)
        v16h a0 = hA[0][lane], a1 = hA[1][lane];
#pragma unroll
        for (int tt = 0; tt < 3; tt++) {
            v8f c = {};
            c = wmma16(a0, bw[tt][0], c);
            c = wmma16(a1, bw[tt][1], c);
            int nb = (wave*3 + tt)*16 + hl;
#pragma unroll
            for (int r = 0; r < 8; r++) gh[g*8 + r][nb] = c[r];
        }
        __syncthreads();
        // phase 2: gates + state update (each thread owns 8 (b,k) cells)
        long row = (long)Bidx * SS + s;
        const float* xpr = xp + row*384 + dir*192;
        f4 xr0 = *(const f4*)(xpr + k0),       xr1 = *(const f4*)(xpr + k0 + 4);
        f4 xz0 = *(const f4*)(xpr + 64 + k0),  xz1 = *(const f4*)(xpr + 64 + k0 + 4);
        f4 xn0 = *(const f4*)(xpr + 128 + k0), xn1 = *(const f4*)(xpr + 128 + k0 + 4);
        float* outp = lstm + row*128 + dir*64 + k0;
#pragma unroll
        for (int i = 0; i < 8; i++) {
            int k = k0 + i;
            float xr = (i < 4) ? xr0[i] : xr1[i-4];
            float xz = (i < 4) ? xz0[i] : xz1[i-4];
            float xn = (i < 4) ? xn0[i] : xn1[i-4];
            float hr = gh[b_loc][k]     + br[i];
            float hz = gh[b_loc][64+k]  + bz[i];
            float hn = gh[b_loc][128+k] + bn[i];
            float rg = 1.f/(1.f + __expf(-(xr + hr)));
            float zg = 1.f/(1.f + __expf(-(xz + hz)));
            float nw = tanhf(xn + rg*hn);
            float hnew = (1.f - zg)*nw + zg*h[i];
            h[i] = hnew;
            outp[i] = hnew;
            // write back into A-fragment order (inverse of pack_a mapping)
            int frag = k >> 5, kk = k & 31;
            int gg = kk >> 4;
            int jj = kk & 15;
            ((_Float16*)&hA[frag][gg*16 + b_loc])[jj] = (_Float16)hnew;
        }
        __syncthreads();
    }
}

// ---------------------------------------------------------------------------
// Kernel 3: qkv = input2 @ in_proj_w^T + in_proj_b.  M=32768, N=96, K=32.
// ---------------------------------------------------------------------------
__global__ void k_qkv(const float* __restrict__ x2, const float* __restrict__ ipw,
                      const float* __restrict__ ipb, float* __restrict__ qkv) {
    int wave = blockIdx.x * 4 + (threadIdx.x >> 5);
    int lane = threadIdx.x & 31, hl = lane & 15, g = lane >> 4;
    int mt = wave / 6, nt = wave % 6;
    const float* arow = x2 + (long)(mt*16 + hl) * 32;
    v16h a = pack_a_from_f32(arow, 0, g);
    int n = nt*16 + hl;
    v16h b = pack_b16(ipw + n*32 + g*16);
    v8f c = {};
    c = wmma16(a, b, c);
    float bv = ipb[n];
#pragma unroll
    for (int r = 0; r < 8; r++)
        qkv[(long)(mt*16 + g*8 + r)*96 + n] = c[r] + bv;
}

// ---------------------------------------------------------------------------
// Kernel 4: attention. One wave per (batch, 16-row q-tile): QK^T (32 WMMA),
// softmax in LDS, P@V (32 WMMA) with probs re-staged as f16 A-fragments.
// ---------------------------------------------------------------------------
__global__ void __launch_bounds__(32) k_attn(const float* __restrict__ qkv,
                                             float* __restrict__ attn) {
    __shared__ float sc[16][512];                         // 32 KB scores
    __shared__ __align__(16) _Float16 pr[16][512];        // 16 KB probs (f16)
    int b = blockIdx.x >> 5, qt = blockIdx.x & 31;
    int lane = threadIdx.x, hl = lane & 15, g = lane >> 4;
    long base = (long)b * SS;

    const float* qrow = qkv + (base + qt*16 + hl) * 96;   // q at offset 0
    v16h aq = pack_a_from_f32(qrow, 0, g);
    const float scale = 0.1767766952966369f;              // 1/sqrt(32)

    for (int kt = 0; kt < 32; kt++) {
        const float* krow = qkv + (base + kt*16 + hl) * 96 + 32;  // k at offset 32
        v16h bk = pack_b16(krow + g*16);
        v8f c = {};
        c = wmma16(aq, bk, c);
#pragma unroll
        for (int r = 0; r < 8; r++) sc[g*8 + r][kt*16 + hl] = c[r] * scale;
    }
    __syncthreads();

    {   // softmax: lane -> (row hl, half g covers 256 keys)
        const float* srow = sc[hl] + g*256;
        float m = -1e30f;
        for (int t = 0; t < 256; t++) m = fmaxf(m, srow[t]);
        m = fmaxf(m, __shfl_xor(m, 16));
        float ss = 0.f;
        for (int t = 0; t < 256; t++) ss += __expf(srow[t] - m);
        ss += __shfl_xor(ss, 16);
        float inv = 1.f / ss;
        _Float16* prow = pr[hl] + g*256;
        for (int t = 0; t < 256; t++) prow[t] = (_Float16)(__expf(srow[t] - m) * inv);
    }
    __syncthreads();

    for (int ntile = 0; ntile < 2; ntile++) {
        v8f c = {};
        int n = ntile*16 + hl;
        for (int kf = 0; kf < 16; kf++) {
            v8h lo = *(const v8h*)&pr[hl][kf*32 + g*16];
            v8h hi = *(const v8h*)&pr[hl][kf*32 + g*16 + 8];
            v16h a;
#pragma unroll
            for (int t = 0; t < 8; t++) { a[t] = lo[t]; a[8+t] = hi[t]; }
            v16h bv;
#pragma unroll
            for (int j = 0; j < 16; j++) {
                int key = kf*32 + g*16 + j;
                bv[j] = (_Float16)qkv[(base + key)*96 + 64 + n];   // v at offset 64
            }
            c = wmma16(a, bv, c);
        }
#pragma unroll
        for (int r = 0; r < 8; r++)
            attn[(base + qt*16 + g*8 + r)*32 + n] = c[r];
    }
}

// ---------------------------------------------------------------------------
// Kernel 5: algebraic fold of dense/out_proj/fusion into combined weights.
// Wc1 = Wdense @ Wfuse[0:16];  Wc2 = out_proj_w^T @ Wfuse[16:48];
// bc  = out_proj_b @ Wfuse[16:48]
// ---------------------------------------------------------------------------
__global__ void k_comb(const float* __restrict__ Wdense, const float* __restrict__ opw,
                       const float* __restrict__ opb, const float* __restrict__ Wfuse,
                       float* __restrict__ Wc1, float* __restrict__ Wc2,
                       float* __restrict__ bc) {
    int tid = threadIdx.x;
    for (int idx = tid; idx < 128*16; idx += blockDim.x) {
        int k = idx >> 4, t = idx & 15;
        float s = 0.f;
        for (int j = 0; j < 16; j++) s += Wdense[k*16 + j] * Wfuse[j*16 + t];
        Wc1[idx] = s;
    }
    for (int idx = tid; idx < 32*16; idx += blockDim.x) {
        int p = idx >> 4, t = idx & 15;
        float s = 0.f;
        for (int j = 0; j < 32; j++) s += opw[j*32 + p] * Wfuse[(16 + j)*16 + t];
        Wc2[idx] = s;
    }
    for (int t = tid; t < 16; t += blockDim.x) {
        float s = 0.f;
        for (int j = 0; j < 32; j++) s += opb[j] * Wfuse[(16 + j)*16 + t];
        bc[t] = s;
    }
}

// ---------------------------------------------------------------------------
// Kernel 6: emissions = lstm_out@Wc1 + attn_raw@Wc2 + bc.  M=32768,N=16,K=160.
// ---------------------------------------------------------------------------
__global__ void k_emis(const float* __restrict__ lstm, const float* __restrict__ attn,
                       const float* __restrict__ Wc1, const float* __restrict__ Wc2,
                       const float* __restrict__ bc, float* __restrict__ emis) {
    int mt = blockIdx.x * 4 + (threadIdx.x >> 5);
    int lane = threadIdx.x & 31, hl = lane & 15, g = lane >> 4;
    const float* arow = lstm + (long)(mt*16 + hl) * 128;
    v8f c = {};
#pragma unroll
    for (int kf = 0; kf < 4; kf++) {
        v16h a = pack_a_from_f32(arow, kf*32, g);
        v16h b;
#pragma unroll
        for (int j = 0; j < 16; j++) b[j] = (_Float16)Wc1[(kf*32 + g*16 + j)*16 + hl];
        c = wmma16(a, b, c);
    }
    {
        const float* arow2 = attn + (long)(mt*16 + hl) * 32;
        v16h a = pack_a_from_f32(arow2, 0, g);
        v16h b;
#pragma unroll
        for (int j = 0; j < 16; j++) b[j] = (_Float16)Wc2[(g*16 + j)*16 + hl];
        c = wmma16(a, b, c);
    }
    float bv = bc[hl];
#pragma unroll
    for (int r = 0; r < 8; r++)
        emis[(long)(mt*16 + g*8 + r)*16 + hl] = c[r] + bv;
}

// ---------------------------------------------------------------------------
// Kernel 7: CRF NLL. 2 batches per wave, lane = (batch-half, tag).
// ---------------------------------------------------------------------------
__global__ void __launch_bounds__(32) k_crf(const float* __restrict__ emis,
                                            const int* __restrict__ targets,
                                            const float* __restrict__ cstart,
                                            const float* __restrict__ cend,
                                            const float* __restrict__ ctrans,
                                            float* __restrict__ lossb) {
    int lane = threadIdx.x;
    int sub = lane >> 4, j = lane & 15;
    int bidx = blockIdx.x * 2 + sub;
    const float* em = emis + (long)bidx * SS * TT;
    const int*   tg = targets + (long)bidx * SS;
    float tr[16];
#pragma unroll
    for (int i = 0; i < 16; i++) tr[i] = ctrans[i*16 + j];

    // numerator
    float np = 0.f;
    for (int t = 0; t < 32; t++) {
        int s = j + 16*t;
        int a = tg[s];
        np += em[s*16 + a];
        if (s < 511) np += ctrans[a*16 + tg[s+1]];
    }
#pragma unroll
    for (int o = 8; o >= 1; o >>= 1) np += __shfl_xor(np, o);
    float numer = np + cstart[tg[0]] + cend[tg[511]];

    // denominator: forward algorithm
    float score = cstart[j] + em[j];
    int basel = sub * 16;
    for (int s = 1; s < SS; s++) {
        float v[16], m = -1e30f;
#pragma unroll
        for (int i = 0; i < 16; i++) {
            v[i] = __shfl(score, basel + i) + tr[i];
            m = fmaxf(m, v[i]);
        }
        float ssum = 0.f;
#pragma unroll
        for (int i = 0; i < 16; i++) ssum += __expf(v[i] - m);
        score = m + __logf(ssum) + em[s*16 + j];
    }
    score += cend[j];
    float m2 = score;
#pragma unroll
    for (int o = 8; o >= 1; o >>= 1) m2 = fmaxf(m2, __shfl_xor(m2, o));
    float e = __expf(score - m2);
#pragma unroll
    for (int o = 8; o >= 1; o >>= 1) e += __shfl_xor(e, o);
    float denom = m2 + __logf(e);
    if (j == 0) lossb[bidx] = denom - numer;
}

// ---------------------------------------------------------------------------
// Kernel 8: mean over 64 batches -> scalar loss
// ---------------------------------------------------------------------------
__global__ void k_reduce(const float* __restrict__ lossb, float* __restrict__ out) {
    if (threadIdx.x == 0) {
        float s = 0.f;
        for (int i = 0; i < 64; i++) s += lossb[i];
        out[0] = s / 64.f;
    }
}

extern "C" void kernel_launch(void* const* d_in, const int* in_sizes, int n_in,
                              void* d_out, int out_size, void* d_ws, size_t ws_size,
                              hipStream_t stream) {
    (void)in_sizes; (void)n_in; (void)out_size; (void)ws_size;
    const float* bert   = (const float*)d_in[0];
    const float* in2    = (const float*)d_in[1];
    const int*   tgt    = (const int*)d_in[2];
    const float* Wih_f  = (const float*)d_in[3];
    const float* Whh_f  = (const float*)d_in[4];
    const float* bih_f  = (const float*)d_in[5];
    const float* bhh_f  = (const float*)d_in[6];
    const float* Wih_b  = (const float*)d_in[7];
    const float* Whh_b  = (const float*)d_in[8];
    const float* bih_b  = (const float*)d_in[9];
    const float* bhh_b  = (const float*)d_in[10];
    const float* Wdense = (const float*)d_in[11];
    const float* ipw    = (const float*)d_in[12];
    const float* ipb    = (const float*)d_in[13];
    const float* opw    = (const float*)d_in[14];
    const float* opb    = (const float*)d_in[15];
    const float* Wfuse  = (const float*)d_in[16];
    const float* cstart = (const float*)d_in[17];
    const float* cend   = (const float*)d_in[18];
    const float* ctrans = (const float*)d_in[19];

    float* ws    = (float*)d_ws;
    float* xp    = ws;                               // 32768*384
    float* lstm  = xp    + (size_t)ROWS*384;         // 32768*128
    float* qkv   = lstm  + (size_t)ROWS*128;         // 32768*96
    float* attn  = qkv   + (size_t)ROWS*96;          // 32768*32
    float* emis  = attn  + (size_t)ROWS*32;          // 32768*16
    float* Wc1   = emis  + (size_t)ROWS*16;          // 2048
    float* Wc2   = Wc1 + 2048;                       // 512
    float* bc    = Wc2 + 512;                        // 16
    float* lossb = bc + 16;                          // 64

    hipLaunchKernelGGL(k_gates,  dim3(12288), dim3(128), 0, stream,
                       bert, Wih_f, Wih_b, bih_f, bih_b, xp);
    hipLaunchKernelGGL(k_qkv,    dim3(3072),  dim3(128), 0, stream, in2, ipw, ipb, qkv);
    hipLaunchKernelGGL(k_gru,    dim3(8),     dim3(128), 0, stream,
                       xp, Whh_f, Whh_b, bhh_f, bhh_b, lstm);
    hipLaunchKernelGGL(k_attn,   dim3(2048),  dim3(32),  0, stream, qkv, attn);
    hipLaunchKernelGGL(k_comb,   dim3(1),     dim3(256), 0, stream,
                       Wdense, opw, opb, Wfuse, Wc1, Wc2, bc);
    hipLaunchKernelGGL(k_emis,   dim3(512),   dim3(128), 0, stream,
                       lstm, attn, Wc1, Wc2, bc, emis);
    hipLaunchKernelGGL(k_crf,    dim3(32),    dim3(32),  0, stream,
                       emis, tgt, cstart, cend, ctrans, lossb);
    hipLaunchKernelGGL(k_reduce, dim3(1),     dim3(32),  0, stream, lossb, (float*)d_out);
}
